// Block_ST_TDC_gra_sharp_2087354106728
// MI455X (gfx1250) — matrix operand order repaired and checked by reading, not compiled, not run on hardware
//
#include <hip/hip_runtime.h>

typedef __attribute__((ext_vector_type(16))) _Float16 v16h;
typedef __attribute__((ext_vector_type(8)))  _Float16 v8h;
typedef __attribute__((ext_vector_type(8)))  float    v8f;

#define F_BN   1
#define F_ELU  2
#define F_BIAS 4
#define F_GS   8

// ---- WMMA fragment index helpers (contiguous k = 16*(lane/16) + e, wave32) ----
__device__ __forceinline__ int ka_idx(int lane, int e) {
  return e + ((lane >> 4) << 4);
}
__device__ __forceinline__ int kb_idx(int lane, int e) {
  return e + ((lane >> 4) << 4);
}

__device__ __forceinline__ v8f wmma_f16(v16h a, v16h b, v8f c) {
  return __builtin_amdgcn_wmma_f32_16x16x32_f16(false, a, false, b, (short)0, c, false, false);
}

// LDS byte-offset of a __shared__ object (ptrtoint of an AS(3) pointer)
typedef __attribute__((address_space(3))) const char lds_char_t;
__device__ __forceinline__ unsigned lds_addr_of(const void* p) {
  return (unsigned)(unsigned long long)(lds_char_t*)p;
}

// =====================================================================
// small utility kernels
// =====================================================================
__global__ void zero_h_kernel(_Float16* p, long n) {
  long i = (long)blockIdx.x * 256 + threadIdx.x;
  if (i < n) p[i] = (_Float16)0.0f;
}

__global__ void pack_mat_kernel(const float* __restrict__ w, _Float16* __restrict__ o, long n) {
  long i = (long)blockIdx.x * 256 + threadIdx.x;
  if (i < n) o[i] = (_Float16)w[i];
}

// CDC-folded weight, TAP-MAJOR K layout: Weff[o][j*192 + c]
__global__ void pack_cdc_kernel(const float* __restrict__ W, _Float16* __restrict__ Weff) {
  int i = blockIdx.x * 256 + threadIdx.x;   // i = o*192 + c
  if (i >= 192 * 192) return;
  int o = i / 192, c = i - o * 192;
  const float* w = W + (long)i * 27;
  float kd = 0.f;
  #pragma unroll
  for (int j = 0; j < 9; ++j) kd += w[j] + w[18 + j];
  _Float16* dst = Weff + (long)o * 5184 + c;
  #pragma unroll
  for (int j = 0; j < 27; ++j) {
    float v = w[j];
    if (j == 13) v -= 0.6f * kd;
    dst[(long)j * 192] = (_Float16)v;
  }
}

__global__ void copy_f32_kernel(const float* __restrict__ s, float* __restrict__ d, long n) {
  long i = (long)blockIdx.x * 256 + threadIdx.x;
  if (i < n) d[i] = s[i];
}

// =====================================================================
// LayerNorm over C=192 -> channel-major f16 hT[c][bp]; optional x copy
// =====================================================================
__global__ void ln_pack_kernel(const float* __restrict__ src, float* __restrict__ xbuf, int copyx,
                               const float* __restrict__ g, const float* __restrict__ b,
                               _Float16* __restrict__ hT) {
  int token = blockIdx.x * 4 + threadIdx.y;   // 0..4095 (= b*2048+p)
  int lane  = threadIdx.x;
  const float* row = src + (long)token * 192;
  float v[6];
  float s = 0.f, ss = 0.f;
  #pragma unroll
  for (int i = 0; i < 6; ++i) {
    float x = row[lane + 32 * i];
    v[i] = x; s += x; ss += x * x;
  }
  #pragma unroll
  for (int m = 1; m < 32; m <<= 1) { s += __shfl_xor(s, m, 32); ss += __shfl_xor(ss, m, 32); }
  float mu   = s * (1.f / 192.f);
  float var  = ss * (1.f / 192.f) - mu * mu;
  float rstd = rsqrtf(var + 1e-6f);
  #pragma unroll
  for (int i = 0; i < 6; ++i) {
    int c = lane + 32 * i;
    hT[(long)c * 4096 + token] = (_Float16)((v[i] - mu) * rstd * g[c] + b[c]);
    if (copyx) xbuf[(long)token * 192 + c] = v[i];
  }
}

// =====================================================================
// CDC conv as implicit-im2col WMMA GEMM, tap-major K.
// 128-thread block = 4 waves x 3 M-tiles; one block per (b,t) frame.
// Per tap: dt/dy/dx uniform, B-frag = 2x ds_load_b128 (shared x-tile in LDS).
// =====================================================================
__global__ void __launch_bounds__(128) conv_cdc_wmma_kernel(
    const _Float16* __restrict__ Weff, const _Float16* __restrict__ hT,
    const float* __restrict__ bg, const float* __restrict__ bb,
    const float* __restrict__ bm, const float* __restrict__ bv,
    _Float16* __restrict__ outH) {
  __shared__ __attribute__((aligned(16))) _Float16 xs[3 * 16 * 192];  // [dt][pos][c]
  __shared__ __attribute__((aligned(16))) _Float16 zrow[192];
  int tid  = threadIdx.x;
  int lane = tid & 31;
  int wid  = tid >> 5;                       // 0..3, handles m-tiles wid*3..wid*3+2
  int nt = blockIdx.x;                       // 0..255
  int b = nt >> 7, t = nt & 127;

  // fill LDS: per (c,dt) pair load one contiguous frame row (32B) and scatter
  for (int i = tid; i < 192 * 3; i += 128) {
    int c = i / 3, dt = i - c * 3;
    int tf = t + dt - 1;
    v8h d0 = {0,0,0,0,0,0,0,0}, d1 = {0,0,0,0,0,0,0,0};
    if (tf >= 0 && tf < 128) {
      const _Float16* sp = hT + (long)c * 4096 + b * 2048 + tf * 16;
      d0 = *(const v8h*)sp;
      d1 = *(const v8h*)(sp + 8);
    }
    _Float16* dst = xs + dt * 3072 + c;
    #pragma unroll
    for (int pos = 0; pos < 8; ++pos) {
      dst[pos * 192]       = d0[pos];
      dst[(pos + 8) * 192] = d1[pos];
    }
  }
  for (int i = tid; i < 192; i += 128) zrow[i] = (_Float16)0.0f;
  __syncthreads();

  int n16 = lane & 15;
  int hh = n16 >> 2, ww = n16 & 3;
  int hi8  = (lane >> 4) << 3;
  int hi16 = (lane >> 4) << 4;
  v8f acc[3];
  #pragma unroll
  for (int m = 0; m < 3; ++m) { v8f z = {0,0,0,0,0,0,0,0}; acc[m] = z; }

  for (int r = 0; r < 27; ++r) {
    int dt = r / 9, r2 = r - dt * 9;
    int dy = r2 / 3, dx = r2 - dy * 3;
    int y = hh + dy - 1, x2 = ww + dx - 1;
    bool valid = ((unsigned)y < 4u) && ((unsigned)x2 < 4u);
    int pos = valid ? (y * 4 + x2) : 0;
    const _Float16* xp = valid ? (xs + dt * 3072 + pos * 192) : zrow;
    long rbase = (long)r * 192;
    #pragma unroll
    for (int cc = 0; cc < 6; ++cc) {
      int c0 = cc * 32;
      v8h b0 = *(const v8h*)(xp + c0 + hi8);
      v8h b1 = *(const v8h*)(xp + c0 + hi8 + 16);
      v16h bf;
      #pragma unroll
      for (int e = 0; e < 8; ++e) { bf[e] = b0[e]; bf[e + 8] = b1[e]; }
      v16h afr[3];
      #pragma unroll
      for (int lmt = 0; lmt < 3; ++lmt) {
        int mtg = wid * 3 + lmt;
        const _Float16* ap = Weff + (long)(mtg * 16 + n16) * 5184 + rbase + c0 + hi8;
        v8h lo = *(const v8h*)ap;
        v8h hc = *(const v8h*)(ap + 16);
        #pragma unroll
        for (int e = 0; e < 8; ++e) { afr[lmt][e] = lo[e]; afr[lmt][e + 8] = hc[e]; }
      }
      #pragma unroll
      for (int lmt = 0; lmt < 3; ++lmt) acc[lmt] = wmma_f16(afr[lmt], bf, acc[lmt]);
    }
  }

  int hi = lane >> 4;
  int p = t * 16 + n16;
  #pragma unroll
  for (int lmt = 0; lmt < 3; ++lmt) {
    #pragma unroll
    for (int r = 0; r < 8; ++r) {
      int o = (wid * 3 + lmt) * 16 + r + 8 * hi;
      float sc  = bg[o] * rsqrtf(bv[o] + 1e-5f);
      float val = (acc[lmt][r] - bm[o]) * sc + bb[o];
      int head = o / 24, dd = o - head * 24;
      outH[((long)(b * 8 + head) * 2048 + p) * 32 + dd] = (_Float16)val;
    }
  }
}

// =====================================================================
// Flash attention: one wave per (b,h, 16-row tile). d padded 24->32.
// K tile (2KB, contiguous) staged via global_load_async_to_lds_b128;
// V is stored d-major (vT[bh][d][p]) so P@V B-frags are contiguous loads.
// =====================================================================
__global__ void __launch_bounds__(32) flash_attn_kernel(
    const _Float16* __restrict__ qh, const _Float16* __restrict__ kh,
    const _Float16* __restrict__ vT, const float* __restrict__ gs,
    _Float16* __restrict__ attT) {
  __shared__ __attribute__((aligned(16))) _Float16 kbuf[32 * 32];
  __shared__ __attribute__((aligned(16))) _Float16 pl[16 * 32];
  int lane = threadIdx.x;
  int mt = blockIdx.x;                 // 0..127
  int bh = blockIdx.y;                 // 0..15
  int b = bh >> 3, h = bh & 7;
  long base  = (long)bh * 2048 * 32;   // qh/kh [bh][p][32]
  long vbase = (long)bh * 32 * 2048;   // vT    [bh][d][p]
  float invgs = 1.0f / gs[0];
  int n16 = lane & 15, hi = lane >> 4;
  int hi8 = hi << 3, hi16 = hi << 4;
  unsigned kb_l = lds_addr_of(kbuf) + lane * 16;

  v16h qa;
  {
    const _Float16* qp = qh + base + (long)(mt * 16 + n16) * 32 + hi8;
    v8h q0 = *(const v8h*)qp;
    v8h q1 = *(const v8h*)(qp + 16);
    #pragma unroll
    for (int e = 0; e < 8; ++e) { qa[e] = q0[e]; qa[e + 8] = q1[e]; }
  }
  float mi[8], li[8];
  #pragma unroll
  for (int r = 0; r < 8; ++r) { mi[r] = -3.0e38f; li[r] = 0.f; }
  v8f acc0 = {0,0,0,0,0,0,0,0}, acc1 = {0,0,0,0,0,0,0,0};

  for (int jt = 0; jt < 64; ++jt) {
    int col0 = jt * 32;
    const char* kg = (const char*)(kh + base + (long)col0 * 32) + lane * 16;
    __syncthreads();   // previous iteration's kbuf reads done before overwrite
    #pragma unroll
    for (int c4 = 0; c4 < 4; ++c4) {
      asm volatile("global_load_async_to_lds_b128 %0, %1, off"
                   :: "v"(kb_l + c4 * 512), "v"((unsigned long long)(kg + c4 * 512)) : "memory");
    }
    asm volatile("s_wait_asynccnt 0x0" ::: "memory");
    __syncthreads();

    v16h kb0, kb1;
    {
      const _Float16* kp0 = kbuf + n16 * 32 + hi8;
      const _Float16* kp1 = kbuf + (16 + n16) * 32 + hi8;
      v8h a0 = *(const v8h*)kp0, a1 = *(const v8h*)(kp0 + 16);
      v8h c0v = *(const v8h*)kp1, c1v = *(const v8h*)(kp1 + 16);
      #pragma unroll
      for (int e = 0; e < 8; ++e) {
        kb0[e] = a0[e]; kb0[e + 8] = a1[e];
        kb1[e] = c0v[e]; kb1[e + 8] = c1v[e];
      }
    }
    v8f z = {0,0,0,0,0,0,0,0};
    v8f s0 = wmma_f16(qa, kb0, z);
    v8f s1 = wmma_f16(qa, kb1, z);
    float p0[8], p1[8];
    #pragma unroll
    for (int r = 0; r < 8; ++r) {
      float a = s0[r] * invgs, c = s1[r] * invgs;
      float mx = fmaxf(a, c);
      #pragma unroll
      for (int m = 1; m < 16; m <<= 1) mx = fmaxf(mx, __shfl_xor(mx, m, 16));
      float nm = fmaxf(mi[r], mx);
      float corr = __expf(mi[r] - nm);
      p0[r] = __expf(a - nm);
      p1[r] = __expf(c - nm);
      float rs = p0[r] + p1[r];
      #pragma unroll
      for (int m = 1; m < 16; m <<= 1) rs += __shfl_xor(rs, m, 16);
      li[r] = li[r] * corr + rs;
      mi[r] = nm;
      acc0[r] *= corr; acc1[r] *= corr;
    }
    __syncthreads();
    #pragma unroll
    for (int r = 0; r < 8; ++r) {
      pl[(r + 8 * hi) * 32 + n16]      = (_Float16)p0[r];
      pl[(r + 8 * hi) * 32 + 16 + n16] = (_Float16)p1[r];
    }
    __syncthreads();
    v16h pa;
    {
      const _Float16* pp = pl + n16 * 32 + hi8;
      v8h a0 = *(const v8h*)pp, a1 = *(const v8h*)(pp + 16);
      #pragma unroll
      for (int e = 0; e < 8; ++e) { pa[e] = a0[e]; pa[e + 8] = a1[e]; }
    }
    // V B-frags: contiguous along k in the d-major layout
    v16h vb0, vb1;
    {
      const _Float16* vp0 = vT + vbase + (long)n16 * 2048 + col0 + hi8;
      const _Float16* vp1 = vT + vbase + (long)(16 + n16) * 2048 + col0 + hi8;
      v8h a0 = *(const v8h*)vp0, a1 = *(const v8h*)(vp0 + 16);
      v8h c0v = *(const v8h*)vp1, c1v = *(const v8h*)(vp1 + 16);
      #pragma unroll
      for (int e = 0; e < 8; ++e) {
        vb0[e] = a0[e]; vb0[e + 8] = a1[e];
        vb1[e] = c0v[e]; vb1[e + 8] = c1v[e];
      }
    }
    acc0 = wmma_f16(pa, vb0, acc0);
    acc1 = wmma_f16(pa, vb1, acc1);
  }
  #pragma unroll
  for (int r = 0; r < 8; ++r) {
    int m = mt * 16 + r + 8 * hi;
    float inv = 1.0f / li[r];
    long bp = (long)b * 2048 + m;
    attT[(long)(h * 24 + n16) * 4096 + bp] = (_Float16)(acc0[r] * inv);
    if (n16 < 8)
      attT[(long)(h * 24 + 16 + n16) * 4096 + bp] = (_Float16)(acc1[r] * inv);
  }
}

// =====================================================================
// Generic batched strided WMMA GEMM with fused epilogues.
// modes: 0 = f16 strided store, 1 = d-major heads store (vT[bh][d][p]),
//        2 = accumulate f32 into x_buf[token][c], 3 = f32 strided store
// =====================================================================
__global__ void __launch_bounds__(32) gemm_wmma_kernel(
    const _Float16* __restrict__ A, long a_sm, long a_sk, long a_b1, long a_b2,
    const _Float16* __restrict__ Bm, long b_sk, long b_sn, long b_b1, long b_b2,
    int z2, int K, int mode, int flags,
    const float* __restrict__ bn_g, const float* __restrict__ bn_b,
    const float* __restrict__ bn_m, const float* __restrict__ bn_v,
    const float* __restrict__ bias, const float* __restrict__ gs,
    float* __restrict__ outf, _Float16* __restrict__ outh,
    long d_sm, long d_sn, long d_b1, long d_b2) {
  int lane = threadIdx.x;
  int mt = blockIdx.x, nt = blockIdx.y, z = blockIdx.z;
  int z1 = z / z2, zr = z - z1 * z2;
  const _Float16* Ab = A  + (long)z1 * a_b1 + (long)zr * a_b2;
  const _Float16* Bb = Bm + (long)z1 * b_b1 + (long)zr * b_b2;
  int n16 = lane & 15, hi = lane >> 4;
  int hi8 = hi << 3, hi16 = hi << 4;
  int m = mt * 16 + n16;
  int n = nt * 16 + n16;
  v8f acc = {0,0,0,0,0,0,0,0};
  for (int k0 = 0; k0 < K; k0 += 32) {
    v16h af, bf;
    if (a_sk == 1) {
      const _Float16* ap = Ab + (long)m * a_sm + k0 + hi16;
      v8h lo = *(const v8h*)ap, hc = *(const v8h*)(ap + 8);
      #pragma unroll
      for (int e = 0; e < 8; ++e) { af[e] = lo[e]; af[e + 8] = hc[e]; }
    } else {
      #pragma unroll
      for (int e = 0; e < 16; ++e)
        af[e] = Ab[(long)m * a_sm + (long)(k0 + ka_idx(lane, e)) * a_sk];
    }
    if (b_sk == 1) {
      const _Float16* bpp = Bb + (long)n * b_sn + k0 + hi16;
      v8h lo = *(const v8h*)bpp, hc = *(const v8h*)(bpp + 8);
      #pragma unroll
      for (int e = 0; e < 8; ++e) { bf[e] = lo[e]; bf[e + 8] = hc[e]; }
    } else {
      #pragma unroll
      for (int e = 0; e < 16; ++e)
        bf[e] = Bb[(long)(k0 + kb_idx(lane, e)) * b_sk + (long)n * b_sn];
    }
    acc = wmma_f16(af, bf, acc);
  }
  float invgs = (flags & F_GS) ? (1.0f / gs[0]) : 1.0f;
  #pragma unroll
  for (int r = 0; r < 8; ++r) {
    int mr = mt * 16 + r + 8 * hi;
    float val = acc[r] * invgs;
    if (flags & F_BN) {
      float s = bn_g[mr] * rsqrtf(bn_v[mr] + 1e-5f);
      val = (val - bn_m[mr]) * s + bn_b[mr];
    }
    if (flags & F_ELU) val = (val > 0.f) ? val : (__expf(val) - 1.0f);
    if (flags & F_BIAS) val += bias[mr];
    if (mode == 0) {
      outh[(long)z1 * d_b1 + (long)zr * d_b2 + (long)mr * d_sm + (long)n * d_sn] = (_Float16)val;
    } else if (mode == 1) {
      int bb2 = n >> 11, p = n & 2047;
      int head = mr / 24, dd = mr - head * 24;
      outh[((long)(bb2 * 8 + head) * 32 + dd) * 2048 + p] = (_Float16)val;
    } else if (mode == 2) {
      outf[(long)n * 192 + mr] += val;
    } else {
      outf[(long)z1 * d_b1 + (long)zr * d_b2 + (long)mr * d_sm + (long)n * d_sn] = val;
    }
  }
}

// =====================================================================
// Depthwise 3x3x3 conv + BN + ELU
// =====================================================================
__global__ void dwconv_kernel(const _Float16* __restrict__ a1T, const float* __restrict__ Wd,
                              const float* __restrict__ g, const float* __restrict__ bb,
                              const float* __restrict__ mm, const float* __restrict__ vv,
                              _Float16* __restrict__ a2T) {
  long idx = (long)blockIdx.x * 256 + threadIdx.x;   // f*4096 + bp
  if (idx >= (long)768 * 4096) return;
  int f = (int)(idx >> 12);
  int bp = (int)(idx & 4095);
  int b = bp >> 11, p = bp & 2047;
  int t = p >> 4, pos = p & 15, hh = pos >> 2, ww = pos & 3;
  const float* w = Wd + (long)f * 27;
  const _Float16* src = a1T + (long)f * 4096 + b * 2048;
  float acc = 0.f;
  #pragma unroll
  for (int dt = 0; dt < 3; ++dt) {
    int tt = t + dt - 1;
    if (tt < 0 || tt >= 128) continue;
    #pragma unroll
    for (int dy = 0; dy < 3; ++dy) {
      int y = hh + dy - 1;
      if (y < 0 || y >= 4) continue;
      #pragma unroll
      for (int dx = 0; dx < 3; ++dx) {
        int x2 = ww + dx - 1;
        if (x2 < 0 || x2 >= 4) continue;
        acc += w[dt * 9 + dy * 3 + dx] * (float)src[tt * 16 + y * 4 + x2];
      }
    }
  }
  float s = g[f] * rsqrtf(vv[f] + 1e-5f);
  float val = (acc - mm[f]) * s + bb[f];
  val = val > 0.f ? val : (__expf(val) - 1.f);
  a2T[idx] = (_Float16)val;
}

// =====================================================================
// Softmax over the HEADS axis of s2 (B,H,C,C); writes f32 output + f16 copy
// =====================================================================
__global__ void softmax_heads_kernel(const float* __restrict__ s2raw,
                                     float* __restrict__ out_s2,
                                     _Float16* __restrict__ s2h) {
  int idx = blockIdx.x * 256 + threadIdx.x;   // b*36864 + c*192 + e
  if (idx >= 2 * 192 * 192) return;
  int b = idx / 36864;
  int ce = idx - b * 36864;
  long base = (long)b * 8 * 36864 + ce;
  float v[8];
  float mx = -3e38f;
  #pragma unroll
  for (int h = 0; h < 8; ++h) { v[h] = s2raw[base + (long)h * 36864]; mx = fmaxf(mx, v[h]); }
  float s = 0.f;
  #pragma unroll
  for (int h = 0; h < 8; ++h) { v[h] = __expf(v[h] - mx); s += v[h]; }
  float inv = 1.f / s;
  #pragma unroll
  for (int h = 0; h < 8; ++h) {
    float p = v[h] * inv;
    out_s2[base + (long)h * 36864] = p;
    s2h[base + (long)h * 36864] = (_Float16)p;
  }
}

// =====================================================================
extern "C" void kernel_launch(void* const* d_in, const int* in_sizes, int n_in,
                              void* d_out, int out_size, void* d_ws, size_t ws_size,
                              hipStream_t stream) {
  (void)in_sizes; (void)n_in; (void)out_size; (void)ws_size;
  const float* x_in   = (const float*)d_in[0];
  const float* Wq     = (const float*)d_in[1];
  const float* bnq_g  = (const float*)d_in[2];
  const float* bnq_b  = (const float*)d_in[3];
  const float* bnq_m  = (const float*)d_in[4];
  const float* bnq_v  = (const float*)d_in[5];
  const float* Wk     = (const float*)d_in[6];
  const float* bnk_g  = (const float*)d_in[7];
  const float* bnk_b  = (const float*)d_in[8];
  const float* bnk_m  = (const float*)d_in[9];
  const float* bnk_v  = (const float*)d_in[10];
  const float* Wv     = (const float*)d_in[11];
  const float* Wqkv2  = (const float*)d_in[12];
  const float* bn2_g  = (const float*)d_in[13];
  const float* bn2_b  = (const float*)d_in[14];
  const float* bn2_m  = (const float*)d_in[15];
  const float* bn2_v  = (const float*)d_in[16];
  const float* Wp     = (const float*)d_in[17];
  const float* bp_    = (const float*)d_in[18];
  const float* ln1_g  = (const float*)d_in[19];
  const float* ln1_b  = (const float*)d_in[20];
  const float* ln2_g  = (const float*)d_in[21];
  const float* ln2_b  = (const float*)d_in[22];
  const float* W1     = (const float*)d_in[23];
  const float* bnf1_g = (const float*)d_in[24];
  const float* bnf1_b = (const float*)d_in[25];
  const float* bnf1_m = (const float*)d_in[26];
  const float* bnf1_v = (const float*)d_in[27];
  const float* Wd     = (const float*)d_in[28];
  const float* bnf2_g = (const float*)d_in[29];
  const float* bnf2_b = (const float*)d_in[30];
  const float* bnf2_m = (const float*)d_in[31];
  const float* bnf2_v = (const float*)d_in[32];
  const float* W2     = (const float*)d_in[33];
  const float* bnf3_g = (const float*)d_in[34];
  const float* bnf3_b = (const float*)d_in[35];
  const float* bnf3_m = (const float*)d_in[36];
  const float* bnf3_v = (const float*)d_in[37];
  const float* gs     = (const float*)d_in[38];
  float* out = (float*)d_out;

  // ---- workspace layout ----
  char* wsb = (char*)d_ws;
  size_t off = 0;
  auto alloc = [&](size_t bytes) -> char* {
    char* p = wsb + off;
    off += (bytes + 255) & ~(size_t)255;
    return p;
  };
  _Float16* hT     = (_Float16*)alloc(192L * 4096 * 2);
  _Float16* Weffq  = (_Float16*)alloc(192L * 5184 * 2);
  _Float16* Weffk  = (_Float16*)alloc(192L * 5184 * 2);
  _Float16* Wv16   = (_Float16*)alloc(36864L * 2);
  _Float16* Wqkv16 = (_Float16*)alloc(36864L * 2);
  _Float16* Wp16   = (_Float16*)alloc(36864L * 2);
  _Float16* W116   = (_Float16*)alloc(147456L * 2);
  _Float16* W216   = (_Float16*)alloc(147456L * 2);
  _Float16* qh     = (_Float16*)alloc(16L * 2048 * 32 * 2);
  _Float16* kh     = (_Float16*)alloc(16L * 2048 * 32 * 2);
  _Float16* vT     = (_Float16*)alloc(16L * 32 * 2048 * 2);
  _Float16* attT   = (_Float16*)alloc(192L * 4096 * 2);
  float*    x_buf  = (float*)   alloc(4096L * 192 * 4);
  _Float16* yT     = (_Float16*)alloc(192L * 4096 * 2);
  float*    s2raw  = (float*)   alloc(589824L * 4);
  _Float16* s2h    = (_Float16*)alloc(589824L * 2);
  _Float16* h2T    = (_Float16*)alloc(192L * 4096 * 2);
  _Float16* a1T    = (_Float16*)alloc(768L * 4096 * 2);
  _Float16* a2T    = (_Float16*)alloc(768L * 4096 * 2);

  const long HBUF = 16L * 2048 * 32;

  // 0) zero q/k/v head buffers (padding lanes must be 0 / finite)
  zero_h_kernel<<<(HBUF + 255) / 256, 256, 0, stream>>>(qh, HBUF);
  zero_h_kernel<<<(HBUF + 255) / 256, 256, 0, stream>>>(kh, HBUF);
  zero_h_kernel<<<(HBUF + 255) / 256, 256, 0, stream>>>(vT, HBUF);

  // 1) weight packing
  pack_cdc_kernel<<<144, 256, 0, stream>>>(Wq, Weffq);
  pack_cdc_kernel<<<144, 256, 0, stream>>>(Wk, Weffk);
  pack_mat_kernel<<<144, 256, 0, stream>>>(Wv, Wv16, 36864);
  pack_mat_kernel<<<144, 256, 0, stream>>>(Wqkv2, Wqkv16, 36864);
  pack_mat_kernel<<<144, 256, 0, stream>>>(Wp, Wp16, 36864);
  pack_mat_kernel<<<576, 256, 0, stream>>>(W1, W116, 147456);
  pack_mat_kernel<<<576, 256, 0, stream>>>(W2, W216, 147456);

  // 2) LN1(x) -> hT; also x_buf = x
  ln_pack_kernel<<<1024, dim3(32, 4), 0, stream>>>(x_in, x_buf, 1, ln1_g, ln1_b, hT);

  // 3) Q = BN(CDC(h)), K = BN(CDC(h))
  conv_cdc_wmma_kernel<<<256, 128, 0, stream>>>(Weffq, hT, bnq_g, bnq_b, bnq_m, bnq_v, qh);
  conv_cdc_wmma_kernel<<<256, 128, 0, stream>>>(Weffk, hT, bnk_g, bnk_b, bnk_m, bnk_v, kh);

  // 4) V = Wv @ hT -> d-major heads layout vT[bh][d][p]
  gemm_wmma_kernel<<<dim3(12, 256, 1), 32, 0, stream>>>(
      Wv16, 192, 1, 0, 0, hT, 4096, 1, 0, 0, 1, 192, 1, 0,
      nullptr, nullptr, nullptr, nullptr, nullptr, nullptr,
      nullptr, vT, 0, 0, 0, 0);

  // 5) fused flash attention -> attT [o][bp]
  flash_attn_kernel<<<dim3(128, 16), 32, 0, stream>>>(qh, kh, vT, gs, attT);

  // 6) x += att @ Wp^T + bp
  gemm_wmma_kernel<<<dim3(12, 256, 1), 32, 0, stream>>>(
      Wp16, 192, 1, 0, 0, attT, 4096, 1, 0, 0, 1, 192, 2, F_BIAS,
      nullptr, nullptr, nullptr, nullptr, bp_, nullptr,
      x_buf, nullptr, 0, 0, 0, 0);

  // 7) LN1(x) again -> hT
  ln_pack_kernel<<<1024, dim3(32, 4), 0, stream>>>(x_buf, x_buf, 0, ln1_g, ln1_b, hT);

  // 8) y = BN2(Wqkv2 @ hT) -> yT [c][bp]
  gemm_wmma_kernel<<<dim3(12, 256, 1), 32, 0, stream>>>(
      Wqkv16, 192, 1, 0, 0, hT, 4096, 1, 0, 0, 1, 192, 0, F_BN,
      bn2_g, bn2_b, bn2_m, bn2_v, nullptr, nullptr,
      nullptr, yT, 4096, 1, 0, 0);

  // 9) s2 = (qc @ qc^T) / gs per (b,h)
  gemm_wmma_kernel<<<dim3(12, 12, 16), 32, 0, stream>>>(
      yT, 4096, 1, 2048, 256, yT, 1, 4096, 2048, 256, 8, 256, 3, F_GS,
      nullptr, nullptr, nullptr, nullptr, nullptr, gs,
      s2raw, nullptr, 192, 1, 294912, 36864);

  // 10) softmax over heads axis -> d_out s2 region + f16 copy
  softmax_heads_kernel<<<288, 256, 0, stream>>>(s2raw, out + 786432, s2h);

  // 11) h2 = s2 @ qc -> h2T [c][bp]
  gemm_wmma_kernel<<<dim3(12, 16, 16), 32, 0, stream>>>(
      s2h, 192, 1, 294912, 36864, yT, 4096, 1, 2048, 256, 8, 192, 0, 0,
      nullptr, nullptr, nullptr, nullptr, nullptr, nullptr,
      nullptr, h2T, 4096, 1, 2048, 256);

  // 12) x += h2 @ Wp^T + bp
  gemm_wmma_kernel<<<dim3(12, 256, 1), 32, 0, stream>>>(
      Wp16, 192, 1, 0, 0, h2T, 4096, 1, 0, 0, 1, 192, 2, F_BIAS,
      nullptr, nullptr, nullptr, nullptr, bp_, nullptr,
      x_buf, nullptr, 0, 0, 0, 0);

  // 13) LN2(x) -> hT
  ln_pack_kernel<<<1024, dim3(32, 4), 0, stream>>>(x_buf, x_buf, 0, ln2_g, ln2_b, hT);

  // 14) a1 = ELU(BN(W1 @ hT)) -> a1T [f][bp]
  gemm_wmma_kernel<<<dim3(48, 256, 1), 32, 0, stream>>>(
      W116, 192, 1, 0, 0, hT, 4096, 1, 0, 0, 1, 192, 0, F_BN | F_ELU,
      bnf1_g, bnf1_b, bnf1_m, bnf1_v, nullptr, nullptr,
      nullptr, a1T, 4096, 1, 0, 0);

  // 15) a2 = ELU(BN(dwconv3x3x3(a1)))
  dwconv_kernel<<<12288, 256, 0, stream>>>(a1T, Wd, bnf2_g, bnf2_b, bnf2_m, bnf2_v, a2T);

  // 16) x += BN(W2 @ a2)
  gemm_wmma_kernel<<<dim3(12, 256, 1), 32, 0, stream>>>(
      W216, 768, 1, 0, 0, a2T, 4096, 1, 0, 0, 1, 768, 2, F_BN,
      bnf3_g, bnf3_b, bnf3_m, bnf3_v, nullptr, nullptr,
      x_buf, nullptr, 0, 0, 0, 0);

  // 17) final x -> d_out
  copy_f32_kernel<<<3072, 256, 0, stream>>>(x_buf, out, 786432);
}
